// SimpleFullDNN_8529805050110
// MI455X (gfx1250) — matrix operand order repaired and checked by reading, not compiled, not run on hardware
//
#include <hip/hip_runtime.h>
#include <hip/hip_bf16.h>

// Problem constants (match reference)
#define P_ 1024
#define K_ 9
#define W_ 10
#define B_ 16384
#define F_ 1024

typedef __attribute__((ext_vector_type(16))) __bf16        v16bf;
typedef __attribute__((ext_vector_type(2)))  __bf16        vbf2;
typedef __attribute__((ext_vector_type(8)))  float         v8f;
typedef __attribute__((ext_vector_type(2)))  float         vf2;
typedef __attribute__((ext_vector_type(8)))  unsigned int  v8u;
typedef __attribute__((ext_vector_type(4)))  unsigned int  v4u;

#define WAVES 8          // waves per block (256 threads, wave32)
#define TILES 2          // 16-row batch tiles per wave per p (amortize weights)
#define PB    64         // patches per block -> gridDim.y = P_/PB = 16
#define HSTR  24         // bf16 row stride for LDS H buffer: 48B rows, 16B aligned

// Fragment as bf16 view + u32 view
union Frag { v8u u; v16bf bf; };

// pack two f32 -> one u32 of two bf16 via vector fptrunc (v_cvt_pk_bf16_f32)
__device__ __forceinline__ unsigned int pk_bf16(float a, float b) {
  vf2  f = {a, b};
  vbf2 h = __builtin_convertvector(f, vbf2);
  union { vbf2 h2; unsigned int u; } z;
  z.h2 = h;
  return z.u;
}

// native v_tanh_f32: __builtin_amdgcn_tanhf is the working name on gfx1250
// (round-2 compile proved it); keep alternates as guarded fallbacks.
__device__ __forceinline__ float fast_tanh(float x) {
#if __has_builtin(__builtin_amdgcn_tanhf)
  return __builtin_amdgcn_tanhf(x);
#elif __has_builtin(__builtin_amdgcn_tanh_f32)
  return __builtin_amdgcn_tanh_f32(x);
#elif __has_builtin(__builtin_amdgcn_exp2f) && __has_builtin(__builtin_amdgcn_rcpf)
  float e = __builtin_amdgcn_exp2f(x * 2.885390081777927f);  // e^(2x)
  return 1.0f - 2.0f * __builtin_amdgcn_rcpf(e + 1.0f);
#else
  float e = __expf(2.0f * x);
  return 1.0f - 2.0f / (e + 1.0f);
#endif
}

__global__ __launch_bounds__(256) void dnn_main(
    const float* __restrict__ x,  const int*   __restrict__ idx,
    const float* __restrict__ W1, const float* __restrict__ b1,
    const float* __restrict__ W2, const float* __restrict__ b2,
    const float* __restrict__ W3, const float* __restrict__ b3,
    const float* __restrict__ wl, float* __restrict__ out)
{
  __shared__ __attribute__((aligned(16))) __bf16 hbuf[WAVES][TILES][16 * HSTR];

  const int lane = threadIdx.x & 31;
  const int wave = threadIdx.x >> 5;
  const int half = lane >> 4;      // 0: lanes 0-15, 1: lanes 16-31
  const int l    = lane & 15;      // row (A) / column (B,D) index
  const int b0   = (blockIdx.x * WAVES + wave) * (16 * TILES);
  const int p0   = blockIdx.y * PB;

  const int  ln   = (l < W_) ? l : 0;  // clamped column for weight loads
  const bool colv = (l < W_);          // live output column
  const bool bv   = (half == 0) && colv;

  float acc[TILES][8];
#pragma unroll
  for (int t = 0; t < TILES; ++t)
#pragma unroll
    for (int r = 0; r < 8; ++r) acc[t][r] = 0.0f;
  float offlane = 0.0f;   // sum_p sum_v b2[v]*W3[v]*wl  (per-column partial)
  float off2    = 0.0f;   // sum_p b3[p]*wl[p]           (lane-uniform)

  v8f c0;
#pragma unroll
  for (int i = 0; i < 8; ++i) c0[i] = 0.0f;

  const float* xr[TILES];
#pragma unroll
  for (int t = 0; t < TILES; ++t)
    xr[t] = x + (size_t)(b0 + 16 * t + l) * F_;

  for (int p = p0; p < p0 + PB; ++p) {
    const float* w1p = W1 + p * (K_ * W_);
    const float* w2p = W2 + p * (W_ * W_);

    // ---- gather columns (lane-uniform) ----
    int c[K_];
#pragma unroll
    for (int j = 0; j < K_; ++j) c[j] = idx[p * K_ + j];

    // ---- B1 fragment: W1[p] as 32x16, lanes 0-15 carry K=0..7, lanes 16-31 K=8 ----
    Frag fb1;
    {
      float e[10];
#pragma unroll
      for (int k = 0; k < K_; ++k) e[k] = (colv && (half ? k < K_ - 8 : k < 8)) ? w1p[(k + 8 * half) * W_ + ln] : 0.0f;
      e[9] = 0.0f;
#pragma unroll
      for (int j = 0; j < 5; ++j) fb1.u[j] = pk_bf16(e[2 * j], e[2 * j + 1]);
#pragma unroll
      for (int j = 5; j < 8; ++j) fb1.u[j] = 0u;
    }
    // ---- B2 fragment: W2[p] as 32x16 ----
    Frag fb2;
    {
      float e[10];
#pragma unroll
      for (int k = 0; k < W_; ++k) e[k] = (colv && (half ? k < W_ - 8 : k < 8)) ? w2p[(k + 8 * half) * W_ + ln] : 0.0f;
#pragma unroll
      for (int j = 0; j < 5; ++j) fb2.u[j] = pk_bf16(e[2 * j], e[2 * j + 1]);
#pragma unroll
      for (int j = 5; j < 8; ++j) fb2.u[j] = 0u;
    }
    // ---- per-column scalars (all lanes; column = l in D layout) ----
    const float b1v = colv ? b1[p * W_ + ln] : 0.0f;
    const float wlp = wl[p];
    const float s3  = (colv ? W3[p * W_ + ln] : 0.0f) * wlp;
    offlane += (colv ? b2[p * W_ + ln] : 0.0f) * s3;
    off2    += b3[p] * wlp;

    // ---- layer 1 for both tiles ----
    v8f d1[TILES];
#pragma unroll
    for (int t = 0; t < TILES; ++t) {
      // A1: lanes 0-15 hold K=0..7 in elems 0..7; lanes 16-31 hold K=8..15.
      Frag a1;
      float e[8];
#pragma unroll
      for (int j = 0; j < 8; ++j) {
        int   col = half ? c[8] : c[j];
        float v   = xr[t][col];
        e[j] = (half && j) ? 0.0f : v;   // f32-side masking
      }
#pragma unroll
      for (int j = 0; j < 4; ++j) a1.u[j] = pk_bf16(e[2 * j], e[2 * j + 1]);
#pragma unroll
      for (int j = 4; j < 8; ++j) a1.u[j] = 0u;

      d1[t] = __builtin_amdgcn_wmma_f32_16x16x32_bf16(
          false, a1.bf, false, fb1.bf, (short)0, c0, false, false);
    }

    // ---- bias + tanh -> LDS (D layout: row = r+8*half, col = l) ----
#pragma unroll
    for (int t = 0; t < TILES; ++t)
#pragma unroll
      for (int r = 0; r < 8; ++r) {
        float h = fast_tanh(d1[t][r] + b1v);
        hbuf[wave][t][(r + 8 * half) * HSTR + l] = (__bf16)h;
      }

    // ---- layer 2 for both tiles ----
#pragma unroll
    for (int t = 0; t < TILES; ++t) {
      // A2: one ds_load_b128 per lane; half 0 -> cols 0..7, half 1 -> cols 8..15
      // (only 8,9 kept). Rows are 48B apart and 16B aligned.
      const __bf16* hrow = &hbuf[wave][t][l * HSTR];
      v4u hv = *(const v4u*)((const char*)hrow + (half ? 16 : 0));
      Frag a2;
      a2.u[0] = hv[0];
      a2.u[1] = half ? 0u : hv[1];
      a2.u[2] = half ? 0u : hv[2];
      a2.u[3] = half ? 0u : hv[3];
#pragma unroll
      for (int j = 4; j < 8; ++j) a2.u[j] = 0u;

      v8f d2 = __builtin_amdgcn_wmma_f32_16x16x32_bf16(
          false, a2.bf, false, fb2.bf, (short)0, c0, false, false);
#pragma unroll
      for (int r = 0; r < 8; ++r) acc[t][r] += d2[r] * s3;
    }
  }

  // ---- reduce across 16 columns (mask<16 keeps halves separate) ----
#pragma unroll
  for (int t = 0; t < TILES; ++t)
#pragma unroll
    for (int r = 0; r < 8; ++r) {
#pragma unroll
      for (int m = 8; m >= 1; m >>= 1) acc[t][r] += __shfl_xor(acc[t][r], m, 32);
    }
#pragma unroll
  for (int m = 8; m >= 1; m >>= 1) offlane += __shfl_xor(offlane, m, 32);

  const float base = offlane + off2;
  if (l == 0) {            // lane 0 -> rows 0..7, lane 16 -> rows 8..15
#pragma unroll
    for (int t = 0; t < TILES; ++t)
#pragma unroll
      for (int r = 0; r < 8; ++r)
        atomicAdd(&out[b0 + 16 * t + 8 * half + r], acc[t][r] + base);
  }
}

__global__ void zero_y(float* __restrict__ out) {
  int i = blockIdx.x * blockDim.x + threadIdx.x;
  if (i < B_) out[i] = 0.0f;
}

__global__ void fill_s(const float* __restrict__ wl, float* __restrict__ out) {
  __shared__ float red[256];
  float s = 0.0f;
  for (int i = threadIdx.x; i < P_; i += 256) s += fabsf(wl[i]);
  red[threadIdx.x] = s;
  __syncthreads();
  for (int st = 128; st > 0; st >>= 1) {
    if ((int)threadIdx.x < st) red[threadIdx.x] += red[threadIdx.x + st];
    __syncthreads();
  }
  const float S = red[0];
  for (int i = threadIdx.x; i < B_; i += 256) out[B_ + i] = S;
}

extern "C" void kernel_launch(void* const* d_in, const int* in_sizes, int n_in,
                              void* d_out, int out_size, void* d_ws, size_t ws_size,
                              hipStream_t stream) {
  (void)in_sizes; (void)n_in; (void)out_size; (void)d_ws; (void)ws_size;
  const float* x   = (const float*)d_in[0];
  const int*   idx = (const int*)  d_in[1];
  const float* W1  = (const float*)d_in[2];
  const float* b1  = (const float*)d_in[3];
  const float* W2  = (const float*)d_in[4];
  const float* b2  = (const float*)d_in[5];
  const float* W3  = (const float*)d_in[6];
  const float* b3  = (const float*)d_in[7];
  const float* wl  = (const float*)d_in[8];
  float* out = (float*)d_out;

  zero_y<<<dim3(B_ / 256), dim3(256), 0, stream>>>(out);
  fill_s<<<dim3(1), dim3(256), 0, stream>>>(wl, out);
  dnn_main<<<dim3(B_ / (16 * TILES * WAVES), P_ / PB), dim3(256), 0, stream>>>(
      x, idx, W1, b1, W2, b2, W3, b3, wl, out);
}
